// GraphMAE_39831526703452
// MI455X (gfx1250) — matrix operand order repaired and checked by reading, not compiled, not run on hardware
//
#include <hip/hip_runtime.h>
#include <hip/hip_bf16.h>

// ---------------------------------------------------------------------------
// GraphMAE forward on gfx1250 (MI455X).
// GEMMs: bf16 WMMA (v_wmma_f32_16x16x32_bf16), f32 accumulate.
// Scatter-add (segment_sum): L2-resident f32 atomics.
// ---------------------------------------------------------------------------

typedef __attribute__((ext_vector_type(16))) __bf16 v16bf;
typedef __attribute__((ext_vector_type(8)))  __bf16 v8bf;
typedef __attribute__((ext_vector_type(8)))  float  v8f;

#define HDIM 128

__device__ __forceinline__ __bf16 f32_to_bf16(float f) {
  unsigned u = __builtin_bit_cast(unsigned, f);
  unsigned r = (u + 0x7FFFu + ((u >> 16) & 1u)) >> 16;   // round-to-nearest-even
  unsigned short s = (unsigned short)r;
  return __builtin_bit_cast(__bf16, s);
}
__device__ __forceinline__ float bf16_to_f32(__bf16 b) {
  unsigned short s = __builtin_bit_cast(unsigned short, b);
  unsigned u = ((unsigned)s) << 16;
  return __builtin_bit_cast(float, u);
}

// --- masking + embed: h = (mask ? token : x) @ W[3,128] + b ----------------
__global__ void k_embed(const float* __restrict__ x,
                        const unsigned char* __restrict__ mask,
                        const float* __restrict__ tok,
                        const float* __restrict__ W,
                        const float* __restrict__ b,
                        float* __restrict__ h, int N) {
  int idx = blockIdx.x * blockDim.x + threadIdx.x;
  int node = idx >> 7, c = idx & 127;
  if (node >= N) return;
  bool m = mask[node] != 0;
  float x0 = m ? tok[0] : x[node * 3 + 0];
  float x1 = m ? tok[1] : x[node * 3 + 1];
  float x2 = m ? tok[2] : x[node * 3 + 2];
  h[idx] = b[c] + x0 * W[c] + x1 * W[128 + c] + x2 * W[256 + c];
}

// --- agg = h (float4 copy) --------------------------------------------------
__global__ void k_copy4(float4* __restrict__ dst, const float4* __restrict__ src, int n4) {
  int i = blockIdx.x * blockDim.x + threadIdx.x;
  if (i < n4) dst[i] = src[i];
}

// --- agg[dst] += h[src]; one wave per edge, lane owns a float4 chunk -------
__global__ void k_scatter(const float* __restrict__ h,
                          const int* __restrict__ src,
                          const int* __restrict__ dst,
                          float* __restrict__ agg, int E) {
  int t = blockIdx.x * blockDim.x + threadIdx.x;
  int e = t >> 5, lane = t & 31;
  if (e >= E) return;
  int s = src[e], d = dst[e];
  float4 v = ((const float4*)(h + (size_t)s * HDIM))[lane];
  float* p = agg + (size_t)d * HDIM + lane * 4;
  atomicAdd(p + 0, v.x);
  atomicAdd(p + 1, v.y);
  atomicAdd(p + 2, v.z);
  atomicAdd(p + 3, v.w);
}

// --- f32 -> bf16 ------------------------------------------------------------
__global__ void k_tobf16(const float* __restrict__ s, __bf16* __restrict__ d, int n) {
  int i = blockIdx.x * blockDim.x + threadIdx.x;
  if (i < n) d[i] = f32_to_bf16(s[i]);
}

// --- repack W[K,Nout] (row-major f32) into WMMA B-fragment order -----------
// P[((nt*(K/32)+kt)*32 + lane)*16 + j] = bf16(W[(kt*32 + ((lane&16)?16:0) + j)*Nout
//                                               + nt*16 + (lane&15)])
__global__ void k_packw(const float* __restrict__ W, __bf16* __restrict__ P,
                        int K, int Nout) {
  int idx = blockIdx.x * blockDim.x + threadIdx.x;
  if (idx >= K * Nout) return;
  int j    = idx & 15;
  int lane = (idx >> 4) & 31;
  int kt   = (idx >> 9) % (K >> 5);
  int nt   = idx / (16 * K);
  int col  = nt * 16 + (lane & 15);
  int kk   = kt * 32 + ((lane & 16) ? 16 : 0) + j;
  P[idx] = f32_to_bf16(W[kk * Nout + col]);
}

// --- GEMM: out[M,Nout] = A[M,K](bf16) * Wpacked + bias; one wave per tile --
template <bool RELU, bool OUTBF>
__global__ __launch_bounds__(256) void k_gemm(const __bf16* __restrict__ A,
                                              const __bf16* __restrict__ P,
                                              const float* __restrict__ bias,
                                              float* __restrict__ outF,
                                              __bf16* __restrict__ outB,
                                              int M, int K, int Nout) {
  int t = blockIdx.x * blockDim.x + threadIdx.x;
  int wave = t >> 5, lane = t & 31;
  int ntiles = Nout >> 4;
  int mt = wave / ntiles;
  int nt = wave - mt * ntiles;
  if (mt >= (M >> 4)) return;          // wave-uniform: EXEC stays all-ones

  int row = mt * 16 + (lane & 15);
  int kb8 = (lane & 16) ? 16 : 0;      // A: lanes 16-31 hold K+16..K+31
  int kts = K >> 5;

  const __bf16* Arow = A + (size_t)row * K;
  const __bf16* Bp   = P + ((size_t)nt * kts * 32 + lane) * 16;

  v8f acc = {0.f, 0.f, 0.f, 0.f, 0.f, 0.f, 0.f, 0.f};
  for (int kt = 0; kt < kts; ++kt) {
    v8bf a0 = *(const v8bf*)(Arow + kt * 32 + kb8);        // K = k0..k0+7   (+kb8)
    v8bf a1 = *(const v8bf*)(Arow + kt * 32 + 8 + kb8);    // K = k0+8..+15  (+kb8)
    v8bf b0 = *(const v8bf*)(Bp + (size_t)kt * 512);
    v8bf b1 = *(const v8bf*)(Bp + (size_t)kt * 512 + 8);
    v16bf av, bv;
#pragma unroll
    for (int i = 0; i < 8; ++i) {
      av[i] = a0[i]; av[i + 8] = a1[i];
      bv[i] = b0[i]; bv[i + 8] = b1[i];
    }
    acc = __builtin_amdgcn_wmma_f32_16x16x32_bf16(
        /*neg_a=*/false, av, /*neg_b=*/false, bv,
        /*c_mod=*/(short)0, acc, /*reuse_a=*/false, /*reuse_b=*/false);
  }

  // C layout: VGPR r -> M = r (lanes 0-15) / 8+r (lanes 16-31); N = lane&15
  int col   = nt * 16 + (lane & 15);
  int rbase = mt * 16 + ((lane & 16) ? 8 : 0);
  float bb = bias[col];
#pragma unroll
  for (int r = 0; r < 8; ++r) {
    float v = acc[r] + bb;
    if (RELU) v = fmaxf(v, 0.f);
    size_t o = (size_t)(rbase + r) * Nout + col;
    if (OUTBF) outB[o] = f32_to_bf16(v);
    else       outF[o] = v;
  }
}

// --- batchnorm stats: stats[c] = sum, stats[128+c] = sumsq -----------------
__global__ void k_bnstats(const float* __restrict__ z, float* __restrict__ stats, int N) {
  int c  = threadIdx.x & 127;
  int ry = threadIdx.x >> 7;              // blockDim 256 -> 2 row-threads
  float s = 0.f, q = 0.f;
  for (int row = blockIdx.x * 2 + ry; row < N; row += gridDim.x * 2) {
    float v = z[(size_t)row * 128 + c];
    s += v; q += v * v;
  }
  atomicAdd(&stats[c], s);
  atomicAdd(&stats[128 + c], q);
}

// --- batchnorm apply + ReLU -------------------------------------------------
__global__ void k_bnapply(const float* __restrict__ z, const float* __restrict__ stats,
                          const float* __restrict__ gamma, const float* __restrict__ beta,
                          float* __restrict__ h, int N) {
  int idx = blockIdx.x * blockDim.x + threadIdx.x;
  if (idx >= N * 128) return;
  int c = idx & 127;
  float invN = 1.0f / (float)N;
  float mean = stats[c] * invN;
  float var  = stats[128 + c] * invN - mean * mean;
  float v = (z[idx] - mean) * rsqrtf(var + 1e-5f) * gamma[c] + beta[c];
  h[idx] = fmaxf(v, 0.f);
}

__global__ void k_zero(float* p, int n) {
  int i = blockIdx.x * blockDim.x + threadIdx.x;
  if (i < n) p[i] = 0.f;
}

// --- decoder second layer (128->3) fused with scaled-cosine loss -----------
__global__ void k_dec2_loss(const __bf16* __restrict__ d1, const float* __restrict__ W2,
                            const float* __restrict__ b2, const float* __restrict__ x,
                            const unsigned char* __restrict__ mask,
                            float* __restrict__ recon, float* __restrict__ accum, int N) {
  int i = blockIdx.x * blockDim.x + threadIdx.x;
  if (i >= N) return;
  float r0 = b2[0], r1 = b2[1], r2 = b2[2];
  const v8bf* dp = (const v8bf*)(d1 + (size_t)i * 128);
#pragma unroll 4
  for (int kb = 0; kb < 16; ++kb) {
    v8bf vv = dp[kb];
#pragma unroll
    for (int u = 0; u < 8; ++u) {
      float a = bf16_to_f32(vv[u]);
      int k = kb * 8 + u;
      r0 += a * W2[k * 3 + 0];
      r1 += a * W2[k * 3 + 1];
      r2 += a * W2[k * 3 + 2];
    }
  }
  recon[i * 3 + 0] = r0; recon[i * 3 + 1] = r1; recon[i * 3 + 2] = r2;
  float x0 = x[i * 3 + 0], x1 = x[i * 3 + 1], x2 = x[i * 3 + 2];
  float nr = fmaxf(sqrtf(r0 * r0 + r1 * r1 + r2 * r2), 1e-12f);
  float nx = fmaxf(sqrtf(x0 * x0 + x1 * x1 + x2 * x2), 1e-12f);
  float cosv = (r0 * x0 + r1 * x1 + r2 * x2) / (nr * nx);
  float e = 1.0f - cosv;
  float pn = e * e;                      // GAMMA = 2
  if (mask[i]) {
    atomicAdd(&accum[0], pn);
    atomicAdd(&accum[1], 1.0f);
  }
}

__global__ void k_loss_final(const float* __restrict__ accum, float* __restrict__ out) {
  out[0] = accum[0] / fmaxf(accum[1], 1.0f);
}

// ---------------------------------------------------------------------------
extern "C" void kernel_launch(void* const* d_in, const int* in_sizes, int n_in,
                              void* d_out, int out_size, void* d_ws, size_t ws_size,
                              hipStream_t stream) {
  const float*         x    = (const float*)d_in[0];
  const int*           ei   = (const int*)d_in[1];
  /* d_in[2] edge_attr: unused by the reference */
  const unsigned char* mask = (const unsigned char*)d_in[3];   // numpy bool = 1 byte
  const float*         tok  = (const float*)d_in[4];
  const float*         eW   = (const float*)d_in[5];
  const float*         eb   = (const float*)d_in[6];
  const float*         cW1  = (const float*)d_in[7];
  const float*         cb1  = (const float*)d_in[8];
  const float*         cW2  = (const float*)d_in[9];
  const float*         cb2  = (const float*)d_in[10];
  const float*         gma  = (const float*)d_in[11];
  const float*         bta  = (const float*)d_in[12];
  const float*         dW1  = (const float*)d_in[13];
  const float*         db1  = (const float*)d_in[14];
  const float*         dW2  = (const float*)d_in[15];
  const float*         db2  = (const float*)d_in[16];

  const int N = in_sizes[0] / 3;
  const int E = in_sizes[1] / 2;
  const int L = 4;
  const int* src = ei;
  const int* dst = ei + E;

  // workspace layout
  char* ws = (char*)d_ws;
  size_t off = 0;
  auto alloc = [&](size_t bytes) -> char* {
    char* p = ws + off;
    off += (bytes + 255) & ~(size_t)255;
    return p;
  };
  float*  h     = (float*)alloc((size_t)N * 128 * 4);
  float*  agg   = (float*)alloc((size_t)N * 128 * 4);   // also reused as z (f32)
  __bf16* actA  = (__bf16*)alloc((size_t)N * 256 * 2);
  __bf16* actB  = (__bf16*)alloc((size_t)N * 256 * 2);
  __bf16* pW1   = (__bf16*)alloc((size_t)L * 128 * 256 * 2);
  __bf16* pW2   = (__bf16*)alloc((size_t)L * 256 * 128 * 2);
  __bf16* pDW   = (__bf16*)alloc((size_t)128 * 128 * 2);
  float*  stats = (float*)alloc(256 * 4);
  float*  accum = (float*)alloc(2 * 4);

  float* loss_out = (float*)d_out;
  float* recon    = loss_out + 1;

  auto cdiv = [](long long a, long long b) { return (int)((a + b - 1) / b); };
  const int B = 256;

  // pack weights to WMMA fragment order (bf16)
  for (int l = 0; l < L; ++l) {
    k_packw<<<cdiv(128 * 256, B), B, 0, stream>>>(cW1 + (size_t)l * 128 * 256,
                                                  pW1 + (size_t)l * 128 * 256, 128, 256);
    k_packw<<<cdiv(256 * 128, B), B, 0, stream>>>(cW2 + (size_t)l * 256 * 128,
                                                  pW2 + (size_t)l * 256 * 128, 256, 128);
  }
  k_packw<<<cdiv(128 * 128, B), B, 0, stream>>>(dW1, pDW, 128, 128);

  // embed
  k_embed<<<cdiv((long long)N * 128, B), B, 0, stream>>>(x, mask, tok, eW, eb, h, N);

  // GIN layers
  for (int l = 0; l < L; ++l) {
    k_copy4<<<cdiv((long long)N * 32, B), B, 0, stream>>>((float4*)agg, (const float4*)h, N * 32);
    k_scatter<<<cdiv((long long)E * 32, B), B, 0, stream>>>(h, src, dst, agg, E);
    k_tobf16<<<cdiv((long long)N * 128, B), B, 0, stream>>>(agg, actA, N * 128);

    // z1 = relu(zin @ W1 + b1)   [N,128]x[128,256] -> bf16
    {
      long long waves = (long long)(N >> 4) * (256 >> 4);
      k_gemm<true, true><<<cdiv(waves * 32, B), B, 0, stream>>>(
          actA, pW1 + (size_t)l * 128 * 256, cb1 + (size_t)l * 256,
          nullptr, actB, N, 128, 256);
    }
    // z = z1 @ W2 + b2           [N,256]x[256,128] -> f32 (into agg)
    {
      long long waves = (long long)(N >> 4) * (128 >> 4);
      k_gemm<false, false><<<cdiv(waves * 32, B), B, 0, stream>>>(
          actB, pW2 + (size_t)l * 256 * 128, cb2 + (size_t)l * 128,
          agg, nullptr, N, 256, 128);
    }
    k_zero<<<1, 256, 0, stream>>>(stats, 256);
    k_bnstats<<<128, 256, 0, stream>>>(agg, stats, N);
    k_bnapply<<<cdiv((long long)N * 128, B), B, 0, stream>>>(
        agg, stats, gma + (size_t)l * 128, bta + (size_t)l * 128, h, N);
  }

  // decoder
  k_tobf16<<<cdiv((long long)N * 128, B), B, 0, stream>>>(h, actA, N * 128);
  {
    long long waves = (long long)(N >> 4) * (128 >> 4);
    k_gemm<true, true><<<cdiv(waves * 32, B), B, 0, stream>>>(
        actA, pDW, db1, nullptr, actB, N, 128, 128);
  }
  k_zero<<<1, 256, 0, stream>>>(accum, 2);
  k_dec2_loss<<<cdiv(N, B), B, 0, stream>>>(actB, dW2, db2, x, mask, recon, accum, N);
  k_loss_final<<<1, 1, 0, stream>>>(accum, loss_out);
}
